// EncoderLayer_13030930776801
// MI455X (gfx1250) — matrix operand order repaired and checked by reading, not compiled, not run on hardware
//
#include <hip/hip_runtime.h>
#include <stdint.h>

typedef __attribute__((ext_vector_type(16))) __bf16 v16bf;
typedef __attribute__((ext_vector_type(8)))  float  v8f;
typedef unsigned short u16;
typedef unsigned int   u32;
typedef long long      i64;

static constexpr int Bc = 2, Sc = 2048, Dc = 1024, Hc = 16, DKc = 64, DVc = 64, DHc = 4096;

__device__ __forceinline__ u16 f2bf(float f) {
  u32 x = __float_as_uint(f);
  u32 r = x + 0x7FFFu + ((x >> 16) & 1u);   // round-to-nearest-even
  return (u16)(r >> 16);
}

union FragU { uint4 q[2]; v16bf v; };

// A: row-major [M, lda] bf16. Fragment = 16x32 tile at (m0, k0).
// ISA 7.12.2: lane L holds row m0+(L&15); VGPR0-3 = K {base..base+7}, VGPR4-7 = K {base+16..base+23}, base = 8*(L>>4).
__device__ __forceinline__ v16bf load_frag_a(const u16* A, int lda, int m0, int k0, int lane) {
  const u16* p = A + (size_t)(m0 + (lane & 15)) * lda + (k0 + ((lane >> 4) << 3));
  FragU f;
  f.q[0] = *(const uint4*)(p);
  f.q[1] = *(const uint4*)(p + 16);
  return f.v;
}

// Bt: row-major [N, ldb] bf16 (i.e. B column-major). Fragment = 32x16 tile at (k0, n0).
// lane L holds col n0+(L&15); VGPR0-3 = K {base..base+7}, VGPR4-7 = K {base+16..base+23}, base = 8*(L>>4).
__device__ __forceinline__ v16bf load_frag_b(const u16* Bt, int ldb, int n0, int k0, int lane) {
  const u16* p = Bt + (size_t)(n0 + (lane & 15)) * ldb + (k0 + ((lane >> 4) << 3));
  FragU f;
  f.q[0] = *(const uint4*)(p);
  f.q[1] = *(const uint4*)(p + 16);
  return f.v;
}

// Generic batched bf16 WMMA GEMM: C = alpha * (A @ Bt^T) + bias + resid, opt ReLU.
// Wave tile 32(M) x 64(N): 2 A frags + 4 B frags -> 8 WMMA per K-step.
// Block = 4 waves = 128(M) x 64(N). Batch offset for X: (z/zdiv)*sXo + (z%zdiv)*sXi.
__global__ void __launch_bounds__(128)
gemm_bf16_wmma(const u16* __restrict__ A, int lda, i64 sAo, i64 sAi,
               const u16* __restrict__ Bt, int ldb, i64 sBo, i64 sBi,
               const float* __restrict__ bias, i64 sbo, i64 sbi,
               const float* __restrict__ resid, int ldr, i64 sro, i64 sri,
               float* __restrict__ outF, int ldf, i64 sfo, i64 sfi,
               u16* __restrict__ outB, int ldg, i64 sgo, i64 sgi,
               int K, int zdiv, float alpha, int relu, int transOutB)
{
  const int z = blockIdx.z;
  const int zo = z / zdiv, zi = z % zdiv;
  A  += zo * sAo + zi * sAi;
  Bt += zo * sBo + zi * sBi;
  if (bias)  bias  += zo * sbo + zi * sbi;
  if (resid) resid += zo * sro + zi * sri;
  if (outF)  outF  += zo * sfo + zi * sfi;
  if (outB)  outB  += zo * sgo + zi * sgi;

  const int lane = threadIdx.x & 31;
  const int wave = threadIdx.x >> 5;
  const int m0 = blockIdx.x * 128 + wave * 32;   // 2 M sub-tiles of 16
  const int n0 = blockIdx.y * 64;

  v8f a0c0 = {}, a0c1 = {}, a0c2 = {}, a0c3 = {};
  v8f a1c0 = {}, a1c1 = {}, a1c2 = {}, a1c3 = {};
  for (int k0 = 0; k0 < K; k0 += 32) {
    v16bf a0 = load_frag_a(A, lda, m0,      k0, lane);
    v16bf a1 = load_frag_a(A, lda, m0 + 16, k0, lane);
    v16bf b0 = load_frag_b(Bt, ldb, n0 +  0, k0, lane);
    v16bf b1 = load_frag_b(Bt, ldb, n0 + 16, k0, lane);
    v16bf b2 = load_frag_b(Bt, ldb, n0 + 32, k0, lane);
    v16bf b3 = load_frag_b(Bt, ldb, n0 + 48, k0, lane);
    a0c0 = __builtin_amdgcn_wmma_f32_16x16x32_bf16(false, a0, false, b0, (short)0, a0c0, false, false);
    a1c0 = __builtin_amdgcn_wmma_f32_16x16x32_bf16(false, a1, false, b0, (short)0, a1c0, false, false);
    a0c1 = __builtin_amdgcn_wmma_f32_16x16x32_bf16(false, a0, false, b1, (short)0, a0c1, false, false);
    a1c1 = __builtin_amdgcn_wmma_f32_16x16x32_bf16(false, a1, false, b1, (short)0, a1c1, false, false);
    a0c2 = __builtin_amdgcn_wmma_f32_16x16x32_bf16(false, a0, false, b2, (short)0, a0c2, false, false);
    a1c2 = __builtin_amdgcn_wmma_f32_16x16x32_bf16(false, a1, false, b2, (short)0, a1c2, false, false);
    a0c3 = __builtin_amdgcn_wmma_f32_16x16x32_bf16(false, a0, false, b3, (short)0, a0c3, false, false);
    a1c3 = __builtin_amdgcn_wmma_f32_16x16x32_bf16(false, a1, false, b3, (short)0, a1c3, false, false);
  }

  // C layout: VGPR r, lane L -> row base + 8*(L>>4) + r, col n0 + 16*t + (L&15)
  const int ncol = lane & 15;
  v8f accs[2][4] = {{a0c0, a0c1, a0c2, a0c3}, {a1c0, a1c1, a1c2, a1c3}};
#pragma unroll
  for (int mt = 0; mt < 2; ++mt) {
    const int mbase = m0 + mt * 16 + ((lane >> 4) << 3);
#pragma unroll
    for (int t = 0; t < 4; ++t) {
      const int n = n0 + t * 16 + ncol;
      const float bv = bias ? bias[n] : 0.f;
      v8f acc = accs[mt][t];
#pragma unroll
      for (int r = 0; r < 8; ++r) {
        const int m = mbase + r;
        float v = acc[r] * alpha + bv;
        if (resid) v += resid[(size_t)m * ldr + n];
        if (relu)  v = v > 0.f ? v : 0.f;
        if (outF)  outF[(size_t)m * ldf + n] = v;
        if (outB) {
          if (transOutB) outB[(size_t)n * ldg + m] = f2bf(v);
          else           outB[(size_t)m * ldg + n] = f2bf(v);
        }
      }
    }
  }
}

__global__ void convert_f32_bf16(const float* __restrict__ in, u16* __restrict__ out, size_t n) {
  size_t i = (size_t)blockIdx.x * blockDim.x + threadIdx.x;
  if (i < n) out[i] = f2bf(in[i]);
}

// in[z][r][c] (f32) -> out[z][c][r] (bf16)
__global__ void transpose_convert(const float* __restrict__ in, u16* __restrict__ out, int rows, int cols) {
  const size_t per = (size_t)rows * cols;
  size_t i = (size_t)blockIdx.x * blockDim.x + threadIdx.x;
  if (i >= per) return;
  const size_t base = (size_t)blockIdx.z * per;
  int r = (int)(i / cols), c = (int)(i % cols);
  out[base + (size_t)c * rows + r] = f2bf(in[base + i]);
}

// Softmax over the QUERY axis: per column s of P[q,s], online max + sum(exp).
__global__ void col_softmax_stats(const float* __restrict__ P, float* __restrict__ cmax,
                                  float* __restrict__ csum, int S) {
  int s = blockIdx.x * blockDim.x + threadIdx.x;
  if (s >= S) return;
  float m = -3.4e38f, sum = 0.f;
  for (int q = 0; q < S; ++q) {
    float x = P[(size_t)q * S + s];
    if (x > m) { sum = sum * expf(m - x) + 1.f; m = x; }
    else       { sum += expf(x - m); }
  }
  cmax[s] = m; csum[s] = sum;
}

__global__ void attn_write_bf16(const float* __restrict__ P, const float* __restrict__ cmax,
                                const float* __restrict__ csum, u16* __restrict__ attn, int S) {
  size_t i = (size_t)blockIdx.x * blockDim.x + threadIdx.x;
  if (i >= (size_t)S * S) return;
  int s = (int)(i & (size_t)(S - 1));
  attn[i] = f2bf(expf(P[i] - cmax[s]) / csum[s]);
}

// LayerNorm, faithful: unbiased std (ddof=1), eps added to std. D fixed = 1024, block = 256.
__global__ void __launch_bounds__(256)
layernorm1024(const float* __restrict__ in, const float* __restrict__ g,
              const float* __restrict__ beta, float* __restrict__ out) {
  const int row = blockIdx.x;
  const float* x = in + (size_t)row * 1024;
  float* y = out + (size_t)row * 1024;
  __shared__ float red[256];
  float loc[4]; float sum = 0.f;
#pragma unroll
  for (int i = 0; i < 4; ++i) { loc[i] = x[threadIdx.x + i * 256]; sum += loc[i]; }
  red[threadIdx.x] = sum; __syncthreads();
  for (int off = 128; off > 0; off >>= 1) {
    if (threadIdx.x < off) red[threadIdx.x] += red[threadIdx.x + off];
    __syncthreads();
  }
  const float mean = red[0] * (1.f / 1024.f);
  __syncthreads();
  float ss = 0.f;
#pragma unroll
  for (int i = 0; i < 4; ++i) { float d = loc[i] - mean; ss += d * d; }
  red[threadIdx.x] = ss; __syncthreads();
  for (int off = 128; off > 0; off >>= 1) {
    if (threadIdx.x < off) red[threadIdx.x] += red[threadIdx.x + off];
    __syncthreads();
  }
  const float stdv = sqrtf(red[0] * (1.f / 1023.f));
  const float inv = 1.f / (stdv + 1e-6f);
#pragma unroll
  for (int i = 0; i < 4; ++i) {
    int c = threadIdx.x + i * 256;
    y[c] = g[c] * (loc[i] - mean) * inv + beta[c];
  }
}

extern "C" void kernel_launch(void* const* d_in, const int* in_sizes, int n_in,
                              void* d_out, int out_size, void* d_ws, size_t ws_size,
                              hipStream_t stream) {
  (void)in_sizes; (void)n_in; (void)out_size; (void)ws_size;
  const float* x   = (const float*)d_in[0];
  const float* Wq  = (const float*)d_in[1];
  const float* bq  = (const float*)d_in[2];
  const float* Wk  = (const float*)d_in[3];
  const float* bk  = (const float*)d_in[4];
  const float* Wv  = (const float*)d_in[5];
  const float* bvp = (const float*)d_in[6];
  const float* Wo  = (const float*)d_in[7];
  const float* bo  = (const float*)d_in[8];
  const float* W1  = (const float*)d_in[9];
  const float* b1  = (const float*)d_in[10];
  const float* W2  = (const float*)d_in[11];
  const float* b2  = (const float*)d_in[12];
  const float* g1  = (const float*)d_in[13];
  const float* be1 = (const float*)d_in[14];
  const float* g2  = (const float*)d_in[15];
  const float* be2 = (const float*)d_in[16];
  float* out = (float*)d_out;

  char* p = (char*)d_ws;
  auto carve = [&](size_t bytes) -> char* {
    char* r = p; p += (bytes + 255) & ~(size_t)255; return r;
  };
  const size_t ND = (size_t)Bc * Sc * Dc;                 // 4M
  u16*   xb    = (u16*)  carve(ND * 2);
  u16*   wqT   = (u16*)  carve((size_t)Hc * DKc * Dc * 2);
  u16*   wkT   = (u16*)  carve((size_t)Hc * DKc * Dc * 2);
  u16*   wvT   = (u16*)  carve((size_t)Hc * DVc * Dc * 2);
  u16*   woT   = (u16*)  carve((size_t)Dc * Hc * DVc * 2);
  u16*   w1T   = (u16*)  carve((size_t)DHc * Dc * 2);
  u16*   w2T   = (u16*)  carve((size_t)Dc * DHc * 2);
  u16*   Qb    = (u16*)  carve((size_t)Bc * Hc * Sc * DKc * 2);
  u16*   Kb    = (u16*)  carve((size_t)Bc * Hc * Sc * DKc * 2);
  u16*   Vt    = (u16*)  carve((size_t)Bc * Hc * DVc * Sc * 2);   // [z][DV][S]
  float* scores= (float*)carve((size_t)Sc * Sc * 4);
  float* cmax  = (float*)carve((size_t)Sc * 4);
  float* csum  = (float*)carve((size_t)Sc * 4);
  u16*   attn  = (u16*)  carve((size_t)Sc * Sc * 2);
  u16*   catV  = (u16*)  carve((size_t)Bc * Sc * Hc * DVc * 2);   // [B,S,H*DV]
  float* res   = (float*)carve(ND * 4);
  u16*   resb  = (u16*)  carve(ND * 2);
  float* sae   = (float*)carve(ND * 4);                           // LN1 output
  u16*   h1    = (u16*)  carve((size_t)Bc * Sc * DHc * 2);
  float* opre  = (float*)carve(ND * 4);

  const dim3 blk256(256), blkG(128);

  // 1) x -> bf16
  convert_f32_bf16<<<dim3((unsigned)((ND + 255) / 256)), blk256, 0, stream>>>(x, xb, ND);
  // 2) weights -> transposed bf16 (Bt layout [N,K])
  transpose_convert<<<dim3((Dc * DKc + 255) / 256, 1, Hc), blk256, 0, stream>>>(Wq, wqT, Dc, DKc);
  transpose_convert<<<dim3((Dc * DKc + 255) / 256, 1, Hc), blk256, 0, stream>>>(Wk, wkT, Dc, DKc);
  transpose_convert<<<dim3((Dc * DVc + 255) / 256, 1, Hc), blk256, 0, stream>>>(Wv, wvT, Dc, DVc);
  transpose_convert<<<dim3((Hc * DVc * Dc + 255) / 256, 1, 1), blk256, 0, stream>>>(Wo, woT, Hc * DVc, Dc);
  transpose_convert<<<dim3((Dc * DHc + 255) / 256, 1, 1), blk256, 0, stream>>>(W1, w1T, Dc, DHc);
  transpose_convert<<<dim3((DHc * Dc + 255) / 256, 1, 1), blk256, 0, stream>>>(W2, w2T, DHc, Dc);

  // 3) Q/K/V projections: batched over z = b*H + h (zdiv = H)
  {
    dim3 g(Sc / 128, DKc / 64, Bc * Hc);
    // Q
    gemm_bf16_wmma<<<g, blkG, 0, stream>>>(
        xb, Dc, (i64)Sc * Dc, 0,
        wqT, Dc, 0, (i64)DKc * Dc,
        bq, 0, DKc,
        nullptr, 0, 0, 0,
        nullptr, 0, 0, 0,
        Qb, DKc, (i64)Hc * Sc * DKc, (i64)Sc * DKc,
        Dc, Hc, 1.0f, 0, 0);
    // K
    gemm_bf16_wmma<<<g, blkG, 0, stream>>>(
        xb, Dc, (i64)Sc * Dc, 0,
        wkT, Dc, 0, (i64)DKc * Dc,
        bk, 0, DKc,
        nullptr, 0, 0, 0,
        nullptr, 0, 0, 0,
        Kb, DKc, (i64)Hc * Sc * DKc, (i64)Sc * DKc,
        Dc, Hc, 1.0f, 0, 0);
    // V, stored transposed: Vt[z][v][s]
    gemm_bf16_wmma<<<g, blkG, 0, stream>>>(
        xb, Dc, (i64)Sc * Dc, 0,
        wvT, Dc, 0, (i64)DVc * Dc,
        bvp, 0, DVc,
        nullptr, 0, 0, 0,
        nullptr, 0, 0, 0,
        Vt, Sc, (i64)Hc * DVc * Sc, (i64)DVc * Sc,
        Dc, Hc, 1.0f, 0, /*transOutB=*/1);
  }

  // 4) Attention per (b,h), reusing one S x S score buffer (bounds workspace)
  for (int z = 0; z < Bc * Hc; ++z) {
    const int b = z / Hc, h = z % Hc;
    // scores[q,s] = (Q K^T)/8, fp32
    gemm_bf16_wmma<<<dim3(Sc / 128, Sc / 64, 1), blkG, 0, stream>>>(
        Qb + (size_t)z * Sc * DKc, DKc, 0, 0,
        Kb + (size_t)z * Sc * DKc, DKc, 0, 0,
        nullptr, 0, 0,
        nullptr, 0, 0, 0,
        scores, Sc, 0, 0,
        nullptr, 0, 0, 0,
        DKc, 1, 0.125f, 0, 0);
    // softmax over the query axis (columns), online max/sum
    col_softmax_stats<<<dim3(Sc / 256), blk256, 0, stream>>>(scores, cmax, csum, Sc);
    attn_write_bf16<<<dim3((unsigned)(((size_t)Sc * Sc + 255) / 256)), blk256, 0, stream>>>(
        scores, cmax, csum, attn, Sc);
    // wV[q,v] = attn @ V  -> write directly into catV[b, q, h*DV + v]
    gemm_bf16_wmma<<<dim3(Sc / 128, DVc / 64, 1), blkG, 0, stream>>>(
        attn, Sc, 0, 0,
        Vt + (size_t)z * DVc * Sc, Sc, 0, 0,
        nullptr, 0, 0,
        nullptr, 0, 0, 0,
        nullptr, 0, 0, 0,
        catV + ((size_t)b * Sc * (Hc * DVc) + (size_t)h * DVc), Hc * DVc, 0, 0,
        Sc, 1, 1.0f, 0, 0);
  }

  // 5) Output projection + residual(x): res = catV @ Wo + bo + x  (fp32 + bf16 copy)
  gemm_bf16_wmma<<<dim3((Bc * Sc) / 128, Dc / 64, 1), blkG, 0, stream>>>(
      catV, Hc * DVc, 0, 0,
      woT, Hc * DVc, 0, 0,
      bo, 0, 0,
      x, Dc, 0, 0,
      res, Dc, 0, 0,
      resb, Dc, 0, 0,
      Hc * DVc, 1, 1.0f, 0, 0);

  // 6) LN1: sae = LN(res)
  layernorm1024<<<dim3(Bc * Sc), blk256, 0, stream>>>(res, g1, be1, sae);

  // 7) FFN1: h1 = relu(res @ W1 + b1), bf16 only
  gemm_bf16_wmma<<<dim3((Bc * Sc) / 128, DHc / 64, 1), blkG, 0, stream>>>(
      resb, Dc, 0, 0,
      w1T, Dc, 0, 0,
      b1, 0, 0,
      nullptr, 0, 0, 0,
      nullptr, 0, 0, 0,
      h1, DHc, 0, 0,
      Dc, 1, 1.0f, /*relu=*/1, 0);

  // 8) FFN2 + residual(sae): opre = h1 @ W2 + b2 + sae
  gemm_bf16_wmma<<<dim3((Bc * Sc) / 128, Dc / 64, 1), blkG, 0, stream>>>(
      h1, DHc, 0, 0,
      w2T, DHc, 0, 0,
      b2, 0, 0,
      sae, Dc, 0, 0,
      opre, Dc, 0, 0,
      nullptr, 0, 0, 0,
      DHc, 1, 1.0f, 0, 0);

  // 9) LN2 -> out
  layernorm1024<<<dim3(Bc * Sc), blk256, 0, stream>>>(opre, g2, be2, out);
}
